// ACTCell_21534966022174
// MI455X (gfx1250) — compile-verified, matrix operand order checked by reading, not run on hardware
//
#include <hip/hip_runtime.h>
#include <hip/hip_bf16.h>

typedef unsigned short bfraw;
typedef __attribute__((ext_vector_type(16))) __bf16 v16bf;
typedef __attribute__((ext_vector_type(8)))  __bf16 v8bf;
typedef __attribute__((ext_vector_type(8)))  float  v8f;

union FragAB { v16bf v; v8bf h[2]; };

#define B_DIM   2048
#define HID     2048
#define OUT_DIM 2048
#define G4H     8192
#define KDIM    2048
#define MAXP    100

#define BM 128
#define BN 256
#define BK 64
#define LDSS 72   // padded LDS row stride (elems); 144B: 16B aligned, conflict-free

__device__ __forceinline__ bfraw f2bf(float f) {
    unsigned u = __float_as_uint(f);
    unsigned r = (u + 0x7FFFu + ((u >> 16) & 1u)) >> 16;  // RNE
    return (bfraw)r;
}
__device__ __forceinline__ float sigf(float x) { return 1.0f / (1.0f + expf(-x)); }

__device__ __forceinline__ unsigned lds_off(const void* p) {
    // generic pointer to __shared__: low 32 bits are the LDS byte offset
    return (unsigned)(unsigned long long)p;
}
// Async DMA: global -> LDS, 16B, tracked with ASYNCcnt (no VGPR round trip)
__device__ __forceinline__ void async_ld_b128(unsigned lds_byte, const bfraw* g) {
    asm volatile("global_load_async_to_lds_b128 %0, %1, off"
                 :: "v"(lds_byte), "v"(g) : "memory");
}
#define WAIT_ASYNC0() asm volatile("s_wait_asynccnt 0" ::: "memory")

// ---------------------------------------------------------------------------
// bf16 WMMA GEMM:  C[M x N] = A[M x K] * B[N x K]^T  (+ epilogue)
//   addend : += addend[idx]           (base0 for gates GEMM)
//   wcol   : -= wcol[col]             (flag-weight column for t>0)
//   bias   : += bias[col], bias2 idem (b_ih+b_hh or b_o)
//   prow   : Cout[idx] += prow[row]*v (ponder-weighted accumulate)
//   ctrl   : early-out when all rows halted
// Block tile 128x256x64, 8 waves (2x4), wave tile 64x64: 32 WMMA per barrier.
// Double-buffered LDS fed by global_load_async_to_lds_b128; 1 barrier/slab.
// ---------------------------------------------------------------------------
__global__ __launch_bounds__(256) void k_gemm(
    const bfraw* __restrict__ A, const bfraw* __restrict__ Bw,
    int N, int K,
    float* __restrict__ Cout,
    const float* __restrict__ addend,
    const float* __restrict__ wcol,
    const float* __restrict__ bias,
    const float* __restrict__ bias2,
    const float* __restrict__ prow,
    const int*  __restrict__ ctrl)
{
    if (ctrl && ctrl[0]) return;   // all rows halted: contributions are zero

    __shared__ bfraw smem[2 * (BM + BN) * LDSS];

    const int tid  = threadIdx.x;
    const int lane = tid & 31;
    const int wave = tid >> 5;
    const int wm   = wave & 1;     // 0..1 -> 64-row strip
    const int wn   = wave >> 1;    // 0..3 -> 64-col strip
    const int tileM = blockIdx.y * BM;
    const int tileN = blockIdx.x * BN;

    const v8f vz = {0.f,0.f,0.f,0.f,0.f,0.f,0.f,0.f};
    v8f acc[4][4];
    #pragma unroll
    for (int i = 0; i < 4; ++i)
        for (int j = 0; j < 4; ++j) acc[i][j] = vz;

    // A staging: 128 rows x 8 chunks = 1024 chunks, 4/thread (64B run)
    const int ar = tid >> 1;
    const int ac = (tid & 1) * 4;        // chunk units (8 elems each)
    // B staging: 256 rows, one full 128B row per thread (8 chunks)
    const unsigned smemBase = lds_off(&smem[0]);
    const unsigned bufBytes = (BM + BN) * LDSS * 2u;
    const unsigned aOffA = (unsigned)(ar * LDSS + ac * 8) * 2u;
    const unsigned aOffB = (unsigned)(BM * LDSS + tid * LDSS) * 2u;

    const bfraw* gA = A  + (size_t)(tileM + ar) * K + ac * 8;
    const bfraw* gB = Bw + (size_t)(tileN + tid) * K;

    auto issue = [&](int b, int k0) {
        const unsigned base = smemBase + (unsigned)b * bufBytes;
        const bfraw* nA = gA + k0;
        const bfraw* nB = gB + k0;
        #pragma unroll
        for (int c = 0; c < 4; ++c)
            async_ld_b128(base + aOffA + 16u * c, nA + 8 * c);
        #pragma unroll
        for (int c = 0; c < 8; ++c)
            async_ld_b128(base + aOffB + 16u * c, nB + 8 * c);
    };

    // fragment addressing per ISA 16-bit layout
    const int mrow = lane & 15;
    const int kb   = (lane >> 4) * 8;

    auto compute = [&](int b) {
        const bfraw* Ab = smem + (size_t)b * (BM + BN) * LDSS;
        const bfraw* Bb = Ab + BM * LDSS;
        #pragma unroll
        for (int kk = 0; kk < 2; ++kk) {
            const int kbb = kk * 32 + kb;
            FragAB bfr[4];
            #pragma unroll
            for (int j = 0; j < 4; ++j) {
                const bfraw* bp = Bb + (wn * 64 + j * 16 + mrow) * LDSS + kbb;
                bfr[j].h[0] = *(const v8bf*)(bp);
                bfr[j].h[1] = *(const v8bf*)(bp + 16);
            }
            #pragma unroll
            for (int i = 0; i < 4; ++i) {
                const bfraw* ap = Ab + (wm * 64 + i * 16 + mrow) * LDSS + kbb;
                FragAB af;
                af.h[0] = *(const v8bf*)(ap);
                af.h[1] = *(const v8bf*)(ap + 16);
                #pragma unroll
                for (int j = 0; j < 4; ++j) {
                    acc[i][j] = __builtin_amdgcn_wmma_f32_16x16x32_bf16(
                        false, af.v, false, bfr[j].v, (short)0, acc[i][j], false, false);
                }
            }
        }
    };

    // ---- software pipeline: 1 barrier per 64-deep slab, last slab peeled
    issue(0, 0);
    int buf = 0;
    for (int k0 = 0; k0 < K - BK; k0 += BK) {
        WAIT_ASYNC0();       // my slab for 'buf' has landed in LDS
        __syncthreads();     // everyone's slab landed; prev compute fully done
        issue(buf ^ 1, k0 + BK);
        compute(buf);
        buf ^= 1;
    }
    WAIT_ASYNC0();
    __syncthreads();
    compute(buf);

    // epilogue: VGPR e -> M=e (lanes 0-15) / M=e+8 (lanes 16-31), N=lane%16
    const int nlo = lane & 15;
    const int mhi = (lane >> 4) * 8;
    #pragma unroll
    for (int i = 0; i < 4; ++i) {
        const int rbase = tileM + wm * 64 + i * 16 + mhi;
        #pragma unroll
        for (int j = 0; j < 4; ++j) {
            const int c = tileN + wn * 64 + j * 16 + nlo;
            float bv = 0.0f;
            if (bias)  bv += bias[c];
            if (bias2) bv += bias2[c];
            if (wcol)  bv -= wcol[c];
            #pragma unroll
            for (int e = 0; e < 8; ++e) {
                const int rr = rbase + e;
                const size_t idx = (size_t)rr * N + c;
                float v = acc[i][j][e] + bv;
                if (addend) v += addend[idx];
                if (prow) Cout[idx] += prow[rr] * v;
                else      Cout[idx]  = v;
            }
        }
    }
}

// ---------------------------------------------------------------------------
// Per-row LSTM elementwise + halting unit. One block per batch row.
// ---------------------------------------------------------------------------
__global__ __launch_bounds__(256) void k_step_elem(
    const float* __restrict__ gates,
    float* __restrict__ cx, bfraw* __restrict__ hx_bf,
    const float* __restrict__ Wp, const float* __restrict__ bp,
    float* __restrict__ gx_acc, float* __restrict__ p_out,
    float* __restrict__ ahx, float* __restrict__ acx,
    int* __restrict__ ctrl)
{
    if (ctrl[0]) return;
    const int row = blockIdx.x;
    const int tid = threadIdx.x;
    __shared__ float red[256];
    __shared__ float p_sh;

    float hv[8], cv[8];
    float dot = 0.0f;
    const size_t gb = (size_t)row * G4H;
    const size_t hb = (size_t)row * HID;
    #pragma unroll
    for (int k = 0; k < 8; ++k) {
        const int c = tid + k * 256;
        const float ig = gates[gb + c];
        const float fg = gates[gb + 2048 + c];
        const float gg = gates[gb + 4096 + c];
        const float og = gates[gb + 6144 + c];
        const float cold = cx[hb + c];
        const float cn = sigf(fg) * cold + sigf(ig) * tanhf(gg);
        const float hn = sigf(og) * tanhf(cn);
        cv[k] = cn; hv[k] = hn;
        cx[hb + c]    = cn;
        hx_bf[hb + c] = f2bf(hn);
        dot += hn * Wp[c];
    }
    red[tid] = dot;
    __syncthreads();
    for (int s = 128; s > 0; s >>= 1) {
        if (tid < s) red[tid] += red[tid + s];
        __syncthreads();
    }
    if (tid == 0) {
        const float gx = sigf(red[0] + bp[0]);
        const float a0 = gx_acc[row];
        const float hp = (a0 > 1.0f - 0.01f) ? 1.0f : 0.0f;
        const float a1 = a0 + gx * (1.0f - hp);
        const float ha = (a1 > 1.0f - 0.01f) ? 1.0f : 0.0f;
        const float halt = ha - hp;
        const float rx = (a1 - 1.0f) * halt;
        const float pv = gx * (1.0f - hp) - rx;
        gx_acc[row] = a1 - rx;
        p_out[row]  = pv;
        p_sh        = pv;
        if (ha < 0.5f) atomicAdd(&ctrl[2], 1);   // count not-yet-halted rows
    }
    __syncthreads();
    const float pv = p_sh;
    #pragma unroll
    for (int k = 0; k < 8; ++k) {
        const int c = tid + k * 256;
        ahx[hb + c] += pv * hv[k];
        acx[hb + c] += pv * cv[k];
    }
}

// ---------------------------------------------------------------------------
// Prep: f32 -> bf16 weight/input conversion + flag-column extraction.
// ---------------------------------------------------------------------------
__global__ void k_prep(const float* __restrict__ Whh, const float* __restrict__ Wo,
                       const float* __restrict__ Wih, const float* __restrict__ x,
                       bfraw* __restrict__ Whh_bf, bfraw* __restrict__ Wo_bf,
                       bfraw* __restrict__ Wih_bf, bfraw* __restrict__ x_bf,
                       float* __restrict__ wflag)
{
    const size_t nW = (size_t)G4H * KDIM;        // 16.78M
    const size_t nS = (size_t)B_DIM * KDIM;      // 4.19M
    for (size_t i = blockIdx.x * blockDim.x + threadIdx.x; i < nW;
         i += (size_t)gridDim.x * blockDim.x) {
        Whh_bf[i] = f2bf(Whh[i]);
        Wih_bf[i] = f2bf(Wih[i]);
        if (i < nS) {
            Wo_bf[i] = f2bf(Wo[i]);
            const int r = (int)(i >> 11), k = (int)(i & 2047);
            x_bf[i] = f2bf(k < 2047 ? x[(size_t)r * 2047 + k] : 1.0f); // flag=1 (step 0)
        }
        if (i < (size_t)G4H) wflag[i] = Wih[i * 2048 + 2047];
    }
}

__global__ void k_init(const float* __restrict__ hx_in, const float* __restrict__ cx_in,
                       float* __restrict__ cx, bfraw* __restrict__ hx_bf,
                       float* __restrict__ gx_acc,
                       float* __restrict__ ahx, float* __restrict__ acx,
                       float* __restrict__ aoutp, int* __restrict__ ctrl)
{
    const size_t n = (size_t)B_DIM * HID;
    for (size_t i = blockIdx.x * blockDim.x + threadIdx.x; i < n;
         i += (size_t)gridDim.x * blockDim.x) {
        ahx[i] = 0.0f; acx[i] = 0.0f; aoutp[i] = 0.0f;
        cx[i] = cx_in[i];
        hx_bf[i] = f2bf(hx_in[i]);
        if (i < (size_t)B_DIM) gx_acc[i] = 0.0f;
        if (i == 0) { ctrl[0] = 0; ctrl[1] = -1; ctrl[2] = 0; }
    }
}

__global__ void k_finalize(int* __restrict__ ctrl, int step)
{
    if (ctrl[0] == 0) {
        if (ctrl[2] == 0) { ctrl[1] = step; ctrl[0] = 1; }  // all halted this step
        ctrl[2] = 0;                                        // reset counter
    }
}

__global__ void k_ponder(const int* __restrict__ ctrl, float* __restrict__ out_last)
{
    const int fh = ctrl[1];
    out_last[0] = (float)(fh >= 0 ? fh + 1 : MAXP);
}

// ---------------------------------------------------------------------------
extern "C" void kernel_launch(void* const* d_in, const int* in_sizes, int n_in,
                              void* d_out, int out_size, void* d_ws, size_t ws_size,
                              hipStream_t stream)
{
    (void)in_sizes; (void)n_in; (void)ws_size;
    const float* x     = (const float*)d_in[0];
    const float* hx_in = (const float*)d_in[1];
    const float* cx_in = (const float*)d_in[2];
    const float* W_ih  = (const float*)d_in[3];
    const float* b_ih  = (const float*)d_in[4];
    const float* W_hh  = (const float*)d_in[5];
    const float* b_hh  = (const float*)d_in[6];
    const float* W_p   = (const float*)d_in[7];
    const float* b_p   = (const float*)d_in[8];
    const float* W_o   = (const float*)d_in[9];
    const float* b_o   = (const float*)d_in[10];

    float* out  = (float*)d_out;
    float* ahx  = out;
    float* acx  = out + (size_t)B_DIM * HID;
    float* aout = out + (size_t)2 * B_DIM * HID;

    char* ws = (char*)d_ws;
    size_t off = 0;
    auto alloc = [&](size_t bytes) -> void* {
        void* p = ws + off;
        off += (bytes + 255) & ~(size_t)255;
        return p;
    };
    bfraw* Whh_bf = (bfraw*)alloc((size_t)G4H * KDIM * 2);     // 32 MB
    bfraw* Wo_bf  = (bfraw*)alloc((size_t)OUT_DIM * HID * 2);  //  8 MB
    float* gates  = (float*)alloc((size_t)B_DIM * G4H * 4);    // 64 MB
    // Wih_bf / x_bf are only live before the first gates GEMM -> alias gates:
    bfraw* Wih_bf = (bfraw*)gates;                                       // 32 MB
    bfraw* x_bf   = (bfraw*)((char*)gates + (size_t)G4H * KDIM * 2);     //  8 MB
    float* base0  = (float*)alloc((size_t)B_DIM * G4H * 4);    // 64 MB
    float* cxw    = (float*)alloc((size_t)B_DIM * HID * 4);    // 16 MB
    bfraw* hx_bf  = (bfraw*)alloc((size_t)B_DIM * HID * 2);    //  8 MB
    float* gx_acc = (float*)alloc((size_t)B_DIM * 4);
    float* p_ws   = (float*)alloc((size_t)B_DIM * 4);
    float* wflag  = (float*)alloc((size_t)G4H * 4);
    int*   ctrl   = (int*)  alloc(256);

    const dim3 blk(256);
    k_prep<<<4096, blk, 0, stream>>>(W_hh, W_o, W_ih, x, Whh_bf, Wo_bf, Wih_bf, x_bf, wflag);
    k_init<<<4096, blk, 0, stream>>>(hx_in, cx_in, cxw, hx_bf, gx_acc, ahx, acx, aout, ctrl);

    // base0 = [x,1] @ W_ih^T + b_ih + b_hh   (flag column included; subtracted for t>0)
    const dim3 gG(G4H / BN, B_DIM / BM);     // (32,16)
    k_gemm<<<gG, blk, 0, stream>>>(x_bf, Wih_bf, G4H, KDIM, base0,
                                   nullptr, nullptr, b_ih, b_hh, nullptr, nullptr);

    const dim3 gO(OUT_DIM / BN, B_DIM / BM); // (8,16)
    for (int t = 0; t < MAXP; ++t) {
        k_gemm<<<gG, blk, 0, stream>>>(hx_bf, Whh_bf, G4H, KDIM, gates,
                                       base0, (t > 0 ? wflag : nullptr),
                                       nullptr, nullptr, nullptr, ctrl);
        k_step_elem<<<B_DIM, blk, 0, stream>>>(gates, cxw, hx_bf, W_p, b_p,
                                               gx_acc, p_ws, ahx, acx, ctrl);
        k_gemm<<<gO, blk, 0, stream>>>(hx_bf, Wo_bf, OUT_DIM, KDIM, aout,
                                       nullptr, nullptr, b_o, nullptr, p_ws, ctrl);
        k_finalize<<<1, 1, 0, stream>>>(ctrl, t);
    }
    k_ponder<<<1, 1, 0, stream>>>(ctrl, out + (size_t)(out_size - 1));
}
